// ConvLayer_51771535786262
// MI455X (gfx1250) — matrix operand order repaired and checked by reading, not compiled, not run on hardware
//
#include <hip/hip_runtime.h>

typedef __attribute__((ext_vector_type(16))) _Float16 v16h;
typedef __attribute__((ext_vector_type(8)))  _Float16 v8h;
typedef __attribute__((ext_vector_type(8)))  float    v8f;
typedef __attribute__((ext_vector_type(4)))  float    v4f;

#define N_NODES 50000
#define N_EDGES 600000

// fast stable softplus: max(x,0) + ln(1+exp(-|x|)); raw v_exp/v_log (safe range)
__device__ __forceinline__ float softplus_f(float x) {
    float t = __builtin_amdgcn_exp2f(-fabsf(x) * 1.4426950408889634f); // t in (0,1]
    float l = __builtin_amdgcn_logf(1.0f + t) * 0.6931471805599453f;   // 1+t in (1,2]
    return fmaxf(x, 0.0f) + l;
}

__device__ __forceinline__ v8f wmma16(v16h a, v16h b, v8f c) {
    return __builtin_amdgcn_wmma_f32_16x16x32_f16(false, a, false, b, (short)0, c, false, false);
}

// load 8 consecutive f32 (16B-aligned) and convert to 8 packed f16
__device__ __forceinline__ v8h load8h(const float* __restrict__ p) {
    const v4f* q = (const v4f*)p;
    v4f a = q[0], b = q[1];
    v8h r;
    r[0] = (_Float16)a[0]; r[1] = (_Float16)a[1]; r[2] = (_Float16)a[2]; r[3] = (_Float16)a[3];
    r[4] = (_Float16)b[0]; r[5] = (_Float16)b[1]; r[6] = (_Float16)b[2]; r[7] = (_Float16)b[3];
    return r;
}

__device__ __forceinline__ const float* edge_src(const float* __restrict__ x,
                                                 const float* __restrict__ ea,
                                                 int r, int c, long e, int K) {
    if (K < 128)      return x  + (long)r * 128 + K;
    else if (K < 256) return x  + (long)c * 128 + (K - 128);
    else              return ea + e * 128 + (K - 256);
}

// ---------------------------------------------------------------------------
__global__ void zero4_kernel(float4* __restrict__ p, long n4) {
    long i = (long)blockIdx.x * blockDim.x + threadIdx.x;
    if (i < n4) p[i] = make_float4(0.f, 0.f, 0.f, 0.f);
}

// ---------------------------------------------------------------------------
// edge MLP + scatter-add.  Block = 8 waves; wave = one 32-edge pair of tiles
// (M=32 blocking: every B fragment is shared by two 16-row WMMAs, halving
// LDS bandwidth per FLOP).  N processed in 4 quarters of 64 cols so the H
// stage stays 2x(16x64) f16 per wave.
// W1 fragments stored t-major (frag = t*12+s) so a tile's 12 K-step
// fragments sit within the 16-bit DS immediate-offset window (s*1KB).
// LDS: W1 swizzled 196608B + W2 swizzled 65536B + 8x4096B stage = 294912B
// ---------------------------------------------------------------------------
__global__ __launch_bounds__(256) void edge_kernel(
    const float* __restrict__ x, const long long* __restrict__ eidx,
    const float* __restrict__ ea,
    const float* __restrict__ W1, const float* __restrict__ b1,
    const float* __restrict__ W2, const float* __restrict__ b2,
    float* __restrict__ aggr)
{
    extern __shared__ _Float16 smem[];
    _Float16* w1s = smem;                    // 98304 halves, fragment-major (t-major)
    _Float16* w2s = smem + 98304;            // 32768 halves, fragment-major
    _Float16* hst = smem + 98304 + 32768;    // 8 waves * 2048 halves

    const int tid = threadIdx.x;

    for (int i = tid; i < 384 * 256; i += 256) {
        int k = i >> 8, n = i & 255;
        int s = k >> 5, kk = k & 31, t = n >> 4, nn = n & 15;
        int ln = nn + (((kk >> 3) & 1) << 4);
        w1s[((t * 12 + s) << 9) + (ln << 4) + ((kk & 7) + ((kk >> 4) << 3))] = (_Float16)W1[i];
    }
    for (int i = tid; i < 256 * 128; i += 256) {
        int k = i >> 7, n = i & 127;
        int s = k >> 5, kk = k & 31, t = n >> 4, nn = n & 15;
        int ln = nn + (((kk >> 3) & 1) << 4);
        w2s[((((s << 3) + t) << 9)) + (ln << 4) + ((kk & 7) + ((kk >> 4) << 3))] = (_Float16)W2[i];
    }
    __syncthreads();

    const int wave = tid >> 5;
    const int lane = tid & 31;
    const int m  = lane & 15;
    const int hi = lane >> 4;
    _Float16* st0 = hst + (wave << 11);      // 16x64 f16 stage, subtile 0
    _Float16* st1 = st0 + 1024;              // 16x64 f16 stage, subtile 1

    const long long* rowI = eidx;
    const long long* colI = eidx + N_EDGES;

    float bias1[16], bias2[8];
#pragma unroll
    for (int t = 0; t < 16; ++t) bias1[t] = b1[(t << 4) + m];
#pragma unroll
    for (int t = 0; t < 8; ++t)  bias2[t] = b2[(t << 4) + m];

    const int npairs = N_EDGES / 32;         // 18750, exact
    for (int pb = blockIdx.x * 8; pb < npairs; pb += gridDim.x * 8) {
        int pair = pb + wave;
        if (pair >= npairs) continue;        // no barriers below: safe divergence
        long ebase = (long)pair * 32;
        long e0 = ebase + m;
        long e1 = ebase + 16 + m;
        int r0 = (int)rowI[e0], c0 = (int)colI[e0];
        int r1 = (int)rowI[e1], c1 = (int)colI[e1];

        int dst0[8], dst1[8];
#pragma unroll
        for (int v = 0; v < 8; ++v) {
            dst0[v] = (int)colI[ebase + v + (hi << 3)];
            dst1[v] = (int)colI[ebase + 16 + v + (hi << 3)];
        }

        // gather A fragments for both 16-edge subtiles (2 x 16x384 f16)
        v16h a1A[12], a1B[12];
#pragma unroll
        for (int s = 0; s < 12; ++s) {
            int g0 = (s << 5) + (hi << 3);
            v8h lo0 = load8h(edge_src(x, ea, r0, c0, e0, g0));
            v8h hh0 = load8h(edge_src(x, ea, r0, c0, e0, g0 + 16));
            a1A[s] = __builtin_shufflevector(lo0, hh0, 0,1,2,3,4,5,6,7,8,9,10,11,12,13,14,15);
            v8h lo1 = load8h(edge_src(x, ea, r1, c1, e1, g0));
            v8h hh1 = load8h(edge_src(x, ea, r1, c1, e1, g0 + 16));
            a1B[s] = __builtin_shufflevector(lo1, hh1, 0,1,2,3,4,5,6,7,8,9,10,11,12,13,14,15);
        }

        v8f acc2A[8] = {}, acc2B[8] = {};

#pragma unroll
        for (int q = 0; q < 4; ++q) {        // 64-column quarters of H
            // ---- layer 1: 4 N-tiles of this quarter; B shared by both subtiles
#pragma unroll
            for (int tt = 0; tt < 4; ++tt) {
                int t = (q << 2) + tt;
                const _Float16* p1 = w1s + ((t * 12) << 9) + (lane << 4);
                v16h bcur = *(const v16h*)p1;
                v8f cA0 = {}, cB0 = {}, cA1 = {}, cB1 = {};
#pragma unroll
                for (int s = 0; s < 12; ++s) {
                    v16h buse = bcur;
                    if (s + 1 < 12) bcur = *(const v16h*)(p1 + ((s + 1) << 9));
                    if (s & 1) { cB0 = wmma16(a1A[s], buse, cB0);
                                 cB1 = wmma16(a1B[s], buse, cB1); }
                    else       { cA0 = wmma16(a1A[s], buse, cA0);
                                 cA1 = wmma16(a1B[s], buse, cA1); }
                }
                float bias = bias1[t];
#pragma unroll
                for (int v = 0; v < 8; ++v) {
                    int haddr = ((v + (hi << 3)) << 6) + (tt << 4) + m;
                    st0[haddr] = (_Float16)softplus_f(cA0[v] + cB0[v] + bias);
                    st1[haddr] = (_Float16)softplus_f(cA1[v] + cB1[v] + bias);
                }
            }
            // ---- layer 2 partial accumulation over this quarter's K range
#pragma unroll
            for (int s2 = 0; s2 < 2; ++s2) {
                int g0 = (s2 << 5) + (hi << 3);
                v8h lo0 = *(const v8h*)(st0 + (m << 6) + g0);
                v8h hh0 = *(const v8h*)(st0 + (m << 6) + g0 + 16);
                v16h a20 = __builtin_shufflevector(lo0, hh0, 0,1,2,3,4,5,6,7,8,9,10,11,12,13,14,15);
                v8h lo1 = *(const v8h*)(st1 + (m << 6) + g0);
                v8h hh1 = *(const v8h*)(st1 + (m << 6) + g0 + 16);
                v16h a21 = __builtin_shufflevector(lo1, hh1, 0,1,2,3,4,5,6,7,8,9,10,11,12,13,14,15);
                int sg = (q << 1) + s2;
                const _Float16* p2 = w2s + (lane << 4) + (sg << 12);
                v16h bcur = *(const v16h*)p2;
#pragma unroll
                for (int t2 = 0; t2 < 8; ++t2) {
                    v16h buse = bcur;
                    if (t2 + 1 < 8) bcur = *(const v16h*)(p2 + ((t2 + 1) << 9));
                    acc2A[t2] = wmma16(a20, buse, acc2A[t2]);
                    acc2B[t2] = wmma16(a21, buse, acc2B[t2]);
                }
            }
        }

        // ---- bias + softplus + scatter-add both subtiles
#pragma unroll
        for (int v = 0; v < 8; ++v) {
            float* ap0 = aggr + (long)dst0[v] * 128 + m;
            float* ap1 = aggr + (long)dst1[v] * 128 + m;
#pragma unroll
            for (int t2 = 0; t2 < 8; ++t2) {
                atomicAdd(ap0 + (t2 << 4), softplus_f(acc2A[t2][v] + bias2[t2]));
                atomicAdd(ap1 + (t2 << 4), softplus_f(acc2B[t2][v] + bias2[t2]));
            }
        }
    }
}

// ---------------------------------------------------------------------------
// node MLP + residual.  Wave = one 16-node tile; Wn swizzled in 64KB LDS.
// ---------------------------------------------------------------------------
__global__ __launch_bounds__(256) void node_kernel(
    const float* __restrict__ x, const float* __restrict__ aggr,
    const float* __restrict__ Wn, const float* __restrict__ bn,
    float* __restrict__ out)
{
    extern __shared__ _Float16 smem[];
    _Float16* wns = smem;
    const int tid = threadIdx.x;
    for (int i = tid; i < 256 * 128; i += 256) {
        int k = i >> 7, n = i & 127;
        int s = k >> 5, kk = k & 31, t = n >> 4, nn = n & 15;
        int ln = nn + (((kk >> 3) & 1) << 4);
        wns[((t * 8 + s) << 9) + (ln << 4) + ((kk & 7) + ((kk >> 4) << 3))] = (_Float16)Wn[i];
    }
    __syncthreads();

    const int wave = tid >> 5;
    const int lane = tid & 31;
    const int m  = lane & 15;
    const int hi = lane >> 4;

    float biasn[8];
#pragma unroll
    for (int t = 0; t < 8; ++t) biasn[t] = bn[(t << 4) + m];

    const int ntiles = N_NODES / 16;         // 3125, exact
    for (int tb = blockIdx.x * 8 + wave; tb < ntiles; tb += gridDim.x * 8) {
        long nbase = (long)tb * 16;
        long node  = nbase + m;
        v16h a[8];
#pragma unroll
        for (int s = 0; s < 8; ++s) {
            int g0 = (s << 5) + (hi << 3);
            int g1 = g0 + 16;
            const float* p0 = (g0 < 128) ? (x + node * 128 + g0) : (aggr + node * 128 + (g0 - 128));
            const float* p1 = (g1 < 128) ? (x + node * 128 + g1) : (aggr + node * 128 + (g1 - 128));
            a[s] = __builtin_shufflevector(load8h(p0), load8h(p1),
                                           0,1,2,3,4,5,6,7,8,9,10,11,12,13,14,15);
        }
#pragma unroll
        for (int t2 = 0; t2 < 8; ++t2) {
            const _Float16* p = wns + ((t2 * 8) << 9) + (lane << 4);
            v16h bcur = *(const v16h*)p;
            v8f cA = {}, cB = {};
#pragma unroll
            for (int s = 0; s < 8; ++s) {
                v16h buse = bcur;
                if (s + 1 < 8) bcur = *(const v16h*)(p + ((s + 1) << 9));
                if (s & 1) cB = wmma16(a[s], buse, cB);
                else       cA = wmma16(a[s], buse, cA);
            }
            float bias = biasn[t2];
#pragma unroll
            for (int v = 0; v < 8; ++v) {
                long row = nbase + v + (hi << 3);
                long off = row * 128 + (t2 << 4) + m;
                out[off] = softplus_f(cA[v] + cB[v] + bias) + x[off];
            }
        }
    }
}

// ---------------------------------------------------------------------------
extern "C" void kernel_launch(void* const* d_in, const int* in_sizes, int n_in,
                              void* d_out, int out_size, void* d_ws, size_t ws_size,
                              hipStream_t stream) {
    const float*     x    = (const float*)d_in[0];
    const long long* eidx = (const long long*)d_in[1];   // int64 [2, E]
    const float*     ea   = (const float*)d_in[2];
    const float*     W1   = (const float*)d_in[3];
    const float*     b1   = (const float*)d_in[4];
    const float*     W2   = (const float*)d_in[5];
    const float*     b2   = (const float*)d_in[6];
    const float*     Wn   = (const float*)d_in[7];
    const float*     bn   = (const float*)d_in[8];
    float* out  = (float*)d_out;
    float* aggr = (float*)d_ws;              // [N_NODES, 128] f32

    long n4 = (long)N_NODES * 128 / 4;
    zero4_kernel<<<(int)((n4 + 255) / 256), 256, 0, stream>>>((float4*)aggr, n4);

    size_t lds_edge = (size_t)(98304 + 32768 + 16384) * sizeof(_Float16); // 294912 B
    edge_kernel<<<1024, 256, lds_edge, stream>>>(x, eidx, ea, W1, b1, W2, b2, aggr);

    size_t lds_node = (size_t)32768 * sizeof(_Float16);                   // 65536 B
    node_kernel<<<391, 256, lds_node, stream>>>(x, aggr, Wn, bn, out);
}
